// SSM2D_66571993088960
// MI455X (gfx1250) — matrix operand order repaired and checked by reading, not compiled, not run on hardware
//
#include <hip/hip_runtime.h>
#include <cstdint>

// ---------------------------------------------------------------------------
// SSM2D on gfx1250 (MI455X):
//  kernel 1 (proj_kernel): dbc = W_dbc @ x  and  delta = softplus(W_dt @ dbc[0:4])
//     via v_wmma_f32_16x16x32_f16 with f16 hi/lo split (3-WMMA ~fp32 emulation).
//     Fully unrolled: named accumulators (no runtime-indexed vector arrays →
//     no v_movrels), ~30 static v_wmma instructions.
//  kernel 2 (scan_kernel): per-(b,c) 2D dyadic pscan fully in LDS (128 KB
//     plane pair in the 320 KB WGP LDS), y accumulated in registers over n.
// ---------------------------------------------------------------------------

typedef __attribute__((ext_vector_type(16))) _Float16 v16h;
typedef __attribute__((ext_vector_type(8)))  float    v8f;

#define HW  16384   // 128*128
#define CCH 64
#define NST 16

static __device__ __forceinline__ v8f wmma_f16(v16h a, v16h b, v8f c) {
  // 8 args: (neg_a, A, neg_b, B, c_mod, C, reuse_a, reuse_b)
  return __builtin_amdgcn_wmma_f32_16x16x32_f16(false, a, false, b, (short)0, c,
                                                false, false);
}

static __device__ __forceinline__ v8f zero8() {
  v8f z;
#pragma unroll
  for (int i = 0; i < 8; ++i) z[i] = 0.0f;
  return z;
}

// Build the A operand (hi/lo split) for output-row block starting at row
// (mt*16 + r), K-step kk.  A layout: element t on lane (r,hi) holds
// K = hi*16+t.
static __device__ __forceinline__ void load_A36(const float* __restrict__ W,
                                                int o, int hi, int kk,
                                                v16h& ah, v16h& al) {
#pragma unroll
  for (int t = 0; t < 16; ++t) {
    int K  = hi * 16 + t;
    int cc = kk * 32 + K;
    float v = (o < 36) ? W[o * 64 + cc] : 0.0f;
    _Float16 h = (_Float16)v;
    ah[t] = h;
    al[t] = (_Float16)(v - (float)h);
  }
}

// ---------------------------------------------------------------------------
// Kernel 1: channel projections.
// One wave32 handles a 16-position tile: D(36x16) = W_dbc(36x64) * x(64x16),
// then delta(64x16) = softplus(W_dt(64x4) * dbc[0:4] + b_dt).
// Operand layouts used here (wave32; A and B share one k order):
//   A element t on lane (r,hi): K = hi*16+t, row M=r
//   B element t on lane (n,hi): K = hi*16+t, col N=n
//   D element v on lane L:      M = v + 8*(L>>4), N = L&15
// ---------------------------------------------------------------------------
__global__ __launch_bounds__(256) void proj_kernel(
    const float* __restrict__ x, const float* __restrict__ W_dbc,
    const float* __restrict__ b_dbc, const float* __restrict__ W_dt,
    const float* __restrict__ b_dt, float* __restrict__ delta_ws,
    float* __restrict__ Bp, float* __restrict__ Cp) {
  const int lane = threadIdx.x & 31;
  const int r = lane & 15;
  const int hi = lane >> 4;
  const int tile = blockIdx.x * 8 + (threadIdx.x >> 5);   // 4096 tiles total
  const int b  = tile >> 11;                              // 2048 tiles / batch
  const int p0 = (tile & 2047) << 4;
  const int p  = p0 + r;
  const float* xB = x + (size_t)b * CCH * HW;

  // ---- B operand (x tile), both K-steps, f16 hi/lo split ----
  v16h bh0, bl0, bh1, bl1;
#pragma unroll
  for (int t = 0; t < 16; ++t) {
    {
      float v = xB[(size_t)(hi * 16 + t) * HW + p];
      _Float16 h = (_Float16)v;
      bh0[t] = h; bl0[t] = (_Float16)(v - (float)h);
    }
    {
      float v = xB[(size_t)(32 + hi * 16 + t) * HW + p];
      _Float16 h = (_Float16)v;
      bh1[t] = h; bl1[t] = (_Float16)(v - (float)h);
    }
  }

  // ---- GEMM 1: dbc = W_dbc @ x, (hi+lo)*(hi+lo) ~= hh + hl + lh ----
  v8f acc0 = zero8(), acc1 = zero8(), acc2 = zero8();
  {
    v16h ah, al;
    // mt = 0
    load_A36(W_dbc, 0 * 16 + r, hi, 0, ah, al);
    acc0 = wmma_f16(ah, bh0, acc0);
    acc0 = wmma_f16(ah, bl0, acc0);
    acc0 = wmma_f16(al, bh0, acc0);
    load_A36(W_dbc, 0 * 16 + r, hi, 1, ah, al);
    acc0 = wmma_f16(ah, bh1, acc0);
    acc0 = wmma_f16(ah, bl1, acc0);
    acc0 = wmma_f16(al, bh1, acc0);
    // mt = 1
    load_A36(W_dbc, 1 * 16 + r, hi, 0, ah, al);
    acc1 = wmma_f16(ah, bh0, acc1);
    acc1 = wmma_f16(ah, bl0, acc1);
    acc1 = wmma_f16(al, bh0, acc1);
    load_A36(W_dbc, 1 * 16 + r, hi, 1, ah, al);
    acc1 = wmma_f16(ah, bh1, acc1);
    acc1 = wmma_f16(ah, bl1, acc1);
    acc1 = wmma_f16(al, bh1, acc1);
    // mt = 2
    load_A36(W_dbc, 2 * 16 + r, hi, 0, ah, al);
    acc2 = wmma_f16(ah, bh0, acc2);
    acc2 = wmma_f16(ah, bl0, acc2);
    acc2 = wmma_f16(al, bh0, acc2);
    load_A36(W_dbc, 2 * 16 + r, hi, 1, ah, al);
    acc2 = wmma_f16(ah, bh1, acc2);
    acc2 = wmma_f16(ah, bl1, acc2);
    acc2 = wmma_f16(al, bh1, acc2);
  }

  // ---- bias (D element v on this lane is row o = mt*16 + hi*8 + v) ----
#pragma unroll
  for (int v = 0; v < 8; ++v) {
    int o0 = 0 * 16 + hi * 8 + v;
    int o1 = 1 * 16 + hi * 8 + v;
    int o2 = 2 * 16 + hi * 8 + v;
    acc0[v] += b_dbc[o0];                       // o0 <= 15, always valid
    acc1[v] += b_dbc[o1];                       // o1 <= 31, always valid
    acc2[v] += (o2 < 36) ? b_dbc[o2] : 0.0f;    // rows 36..47 are padding
  }

  // ---- GEMM 2: delta = softplus(W_dt @ dbc[0:4] + b_dt) ----
  // B operand rows K=0..3 are dbc rows 0..3 = acc0[0..3] on lanes hi==0
  // (B layout wants K = hi*16 + t, so only hi==0, t<4 are nonzero). In-register.
  v16h b2h, b2l;
#pragma unroll
  for (int t = 0; t < 16; ++t) { b2h[t] = (_Float16)0.0f; b2l[t] = (_Float16)0.0f; }
#pragma unroll
  for (int t = 0; t < 4; ++t) {
    float v = (hi == 0) ? acc0[t] : 0.0f;
    _Float16 h = (_Float16)v;
    b2h[t] = h;
    b2l[t] = (_Float16)(v - (float)h);
  }
#pragma unroll
  for (int mt2 = 0; mt2 < 4; ++mt2) {
    v16h a2h, a2l;
#pragma unroll
    for (int t = 0; t < 16; ++t) { a2h[t] = (_Float16)0.0f; a2l[t] = (_Float16)0.0f; }
#pragma unroll
    for (int t = 0; t < 4; ++t) {            // A layout: K<4 only at hi==0, t<4
      float v = (hi == 0) ? W_dt[(mt2 * 16 + r) * 4 + t] : 0.0f;
      _Float16 h = (_Float16)v;
      a2h[t] = h;
      a2l[t] = (_Float16)(v - (float)h);
    }
    v8f d = zero8();
    d = wmma_f16(a2h, b2h, d);
    d = wmma_f16(a2h, b2l, d);
    d = wmma_f16(a2l, b2h, d);
#pragma unroll
    for (int v = 0; v < 8; ++v) {
      int c = mt2 * 16 + hi * 8 + v;
      float z  = d[v] + b_dt[c];
      float sp = fmaxf(z, 0.0f) + log1pf(expf(-fabsf(z)));   // softplus
      delta_ws[(((size_t)b * CCH + c) << 14) + p] = sp;
    }
  }

  // ---- scatter B (rows 4..19) and C (rows 20..35) planes ----
#pragma unroll
  for (int v = 0; v < 8; ++v) {
    int o0 = hi * 8 + v;            // from acc0: rows 0..15
    if (o0 >= 4)
      Bp[(((size_t)b * NST + (o0 - 4)) << 14) + p] = acc0[v];
    int o1 = 16 + hi * 8 + v;       // from acc1: rows 16..31
    if (o1 < 20)
      Bp[(((size_t)b * NST + (o1 - 4)) << 14) + p] = acc1[v];
    else
      Cp[(((size_t)b * NST + (o1 - 20)) << 14) + p] = acc1[v];
    int o2 = 32 + hi * 8 + v;       // from acc2: rows 32..47 (36+ padding)
    if (o2 < 36)
      Cp[(((size_t)b * NST + (o2 - 20)) << 14) + p] = acc2[v];
  }
}

// ---------------------------------------------------------------------------
// Kernel 2: one workgroup (1024 thr) per (b,c). For each n: build A/X planes
// in LDS, run the 4 dyadic pscan phases (reference-exact index pattern),
// accumulate y in registers. 128 KB LDS per workgroup (A+X planes).
// ---------------------------------------------------------------------------
__global__ __launch_bounds__(1024) void scan_kernel(
    const float* __restrict__ x, const float* __restrict__ A_log,
    const float* __restrict__ D_param, const float* __restrict__ delta_ws,
    const float* __restrict__ Bp, const float* __restrict__ Cp,
    float* __restrict__ out) {
  extern __shared__ float smem[];
  float* Al = smem;        // 16384 floats
  float* Xl = smem + HW;   // 16384 floats

  const int bc  = blockIdx.x;
  const int b   = bc >> 6;
  const int c   = bc & 63;
  const int tid = threadIdx.x;
  const float* dpl = delta_ws + (size_t)bc * HW;
  const float* xpl = x + (size_t)bc * HW;

  float yacc[16];
#pragma unroll
  for (int i = 0; i < 16; ++i) yacc[i] = 0.0f;

  for (int n = 0; n < NST; ++n) {
    const float Acn = -__expf(A_log[c * NST + n]);
    const float* bpl = Bp + (((size_t)b * NST + n) << 14);
    const float* cpl = Cp + (((size_t)b * NST + n) << 14);

    // elementwise: A_bar = exp(delta*A), BX = delta*B*x
#pragma unroll
    for (int i = 0; i < 16; ++i) {
      int p = i * 1024 + tid;
      __builtin_prefetch(cpl + p, 0, 0);   // warm Cm plane for y-accum
      float dv = dpl[p];
      float bv = bpl[p];
      float xv = xpl[p];
      Al[p] = __expf(dv * Acn);
      Xl[p] = dv * bv * xv;
    }
    __syncthreads();

    // Phase 1: W up-sweep (targets ≡ -1 mod 2s, sources ≡ s-1 → disjoint)
    for (int k = 0; k < 7; ++k) {
      int per_row = 64 >> k;
      int tot = 128 * per_row;
      for (int u = tid; u < tot; u += 1024) {
        int row  = u >> (6 - k);
        int slot = u & (per_row - 1);
        int tcol = ((slot + 1) << (k + 1)) - 1;
        int scol = tcol - (1 << k);
        int base = row << 7;
        float a  = Al[base + tcol];
        Xl[base + tcol] += a * Xl[base + scol];
        Al[base + tcol]  = a * Al[base + scol];
      }
      __syncthreads();
    }
    // Phase 2: H up-sweep on last column (W-1)
    for (int k = 0; k < 7; ++k) {
      int cnt = 64 >> k;
      if (tid < cnt) {
        int trow = ((tid + 1) << (k + 1)) - 1;
        int srow = trow - (1 << k);
        int ti = (trow << 7) + 127, si = (srow << 7) + 127;
        float a = Al[ti];
        Xl[ti] += a * Xl[si];
        Al[ti]  = a * Al[si];
      }
      __syncthreads();
    }
    // Phase 3: H down-sweep, full rows (rows 3s-1::2s += A * rows 2s-1::2s)
    for (int k = 6; k >= 0; --k) {
      int s   = 1 << k;
      int cnt = (128 - s) >> (k + 1);
      int tot = cnt << 7;
      for (int u = tid; u < tot; u += 1024) {
        int m = u >> 7, j = u & 127;
        int trow = ((2 * m + 3) << k) - 1;
        int srow = trow - s;
        int ti = (trow << 7) + j, si = (srow << 7) + j;
        float a = Al[ti];
        Xl[ti] += a * Xl[si];
        Al[ti]  = a * Al[si];
      }
      __syncthreads();
    }
    // Phase 4: W down-sweep
    for (int k = 6; k >= 0; --k) {
      int s   = 1 << k;
      int cnt = (128 - s) >> (k + 1);
      int tot = cnt << 7;
      for (int u = tid; u < tot; u += 1024) {
        int m = u >> 7, row = u & 127;
        int tcol = ((2 * m + 3) << k) - 1;
        int scol = tcol - s;
        int base = row << 7;
        float a  = Al[base + tcol];
        Xl[base + tcol] += a * Xl[base + scol];
        Al[base + tcol]  = a * Al[base + scol];
      }
      __syncthreads();
    }

    // y[p] += hs[p] * Cm[p]  (thread owns fixed positions across all n)
#pragma unroll
    for (int i = 0; i < 16; ++i) {
      int p = i * 1024 + tid;
      yacc[i] += Xl[p] * cpl[p];
    }
    __syncthreads();   // protect Xl/Al before next n overwrites them
  }

  const float Dv = D_param[c];
#pragma unroll
  for (int i = 0; i < 16; ++i) {
    int p = i * 1024 + tid;
    out[(((size_t)b * HW + p) << 6) + c] = yacc[i] + Dv * xpl[p];
  }
}

// ---------------------------------------------------------------------------
extern "C" void kernel_launch(void* const* d_in, const int* in_sizes, int n_in,
                              void* d_out, int out_size, void* d_ws, size_t ws_size,
                              hipStream_t stream) {
  const float* x      = (const float*)d_in[0];
  const float* W_dbc  = (const float*)d_in[1];
  const float* b_dbc  = (const float*)d_in[2];
  const float* W_dt   = (const float*)d_in[3];
  const float* b_dt   = (const float*)d_in[4];
  const float* A_log  = (const float*)d_in[5];
  const float* D_par  = (const float*)d_in[6];
  float* out = (float*)d_out;

  // workspace: delta planes [b][c][hw] (8 MB) + B planes [b][n][hw] (2 MB)
  //          + C planes [b][n][hw] (2 MB)  => 12 MB total
  float* delta_ws = (float*)d_ws;
  float* Bp = delta_ws + (size_t)2 * CCH * HW;
  float* Cp = Bp + (size_t)2 * NST * HW;

  // 4096 position-tiles (16 positions each), 8 waves per block
  proj_kernel<<<512, 256, 0, stream>>>(x, W_dbc, b_dbc, W_dt, b_dt,
                                       delta_ws, Bp, Cp);
  // one workgroup per (b,c); 128 KB dynamic LDS (A+X planes)
  scan_kernel<<<128, 1024, 2 * HW * sizeof(float), stream>>>(
      x, A_log, D_par, delta_ws, Bp, Cp, out);
}
